// ResnetRNNFPV_25881472926544
// MI455X (gfx1250) — matrix operand order repaired and checked by reading, not compiled, not run on hardware
//
#include <hip/hip_runtime.h>

typedef unsigned short u16;
typedef unsigned int   u32;
typedef __attribute__((ext_vector_type(16))) __bf16 v16bf;
typedef __attribute__((ext_vector_type(8)))  float  v8f;

__device__ __forceinline__ u16 f2bf(float f) {
  u32 u = __float_as_uint(f);
  u32 r = u + 0x7FFFu + ((u >> 16) & 1u);   // round-to-nearest-even
  return (u16)(r >> 16);
}
__device__ __forceinline__ float bf2f(u16 h) {
  return __uint_as_float(((u32)h) << 16);
}

// ---------------------------------------------------------------------------
// im2col: (B,Cin,IH,IW) -> A[M x Kp] bf16, M = B*OH*OW, K index = ci*KH*KH+kh*KH+kw
// ---------------------------------------------------------------------------
template <int SRCF32>
__global__ void im2col_kernel(const void* __restrict__ src, u16* __restrict__ A,
                              int Cin, int IH, int IW, int OH, int OW, int KH,
                              int stride, int pad, int Kreal, int Kp, long long total)
{
  long long idx = (long long)blockIdx.x * blockDim.x + threadIdx.x;
  if (idx >= total) return;
  int k = (int)(idx % Kp);
  long long m = idx / Kp;
  u16 v = 0;
  if (k < Kreal) {
    int kk2 = KH * KH;
    int ci  = k / kk2;
    int rem = k - ci * kk2;
    int kh  = rem / KH;
    int kw  = rem - kh * KH;
    int ow  = (int)(m % OW);
    long long t = m / OW;
    int oh  = (int)(t % OH);
    int b   = (int)(t / OH);
    int ih  = oh * stride - pad + kh;
    int iw  = ow * stride - pad + kw;
    if (ih >= 0 && ih < IH && iw >= 0 && iw < IW) {
      long long off = (((long long)b * Cin + ci) * IH + ih) * IW + iw;
      float x = SRCF32 ? ((const float*)src)[off] : bf2f(((const u16*)src)[off]);
      v = f2bf(x);
    }
  }
  A[idx] = v;
}

// OIHW fp32 weights -> Bmat[N x Kp] bf16 row-major (zero-pad K)
__global__ void pack_w_kernel(const float* __restrict__ W, u16* __restrict__ Bm,
                              int Kreal, int Kp, long long total)
{
  long long idx = (long long)blockIdx.x * blockDim.x + threadIdx.x;
  if (idx >= total) return;
  int k = (int)(idx % Kp);
  long long n = idx / Kp;
  Bm[idx] = (k < Kreal) ? f2bf(W[n * Kreal + k]) : (u16)0;
}

// ---------------------------------------------------------------------------
// WMMA bf16 GEMM: C[M,N] = A[M,K] * B[N,K]^T, fused bias+relu, NCHW bf16 out.
// Block 256 thr = 8 waves; block tile 128x128; wave tile 64x32 (4x2 wmma accs).
// Tiles staged with CDNA5 async global->LDS DMA (ASYNCcnt), double buffered.
// ---------------------------------------------------------------------------
#define KTP 40   // LDS row stride (ushorts): 80B -> bank-conflict-free, 16B aligned

struct Frag16 { union { v16bf v; uint4 q[2]; }; };

#define WAIT_ASYNC(n) asm volatile("s_wait_asynccnt %0" :: "n"(n) : "memory")

__global__ __launch_bounds__(256)
void wmma_gemm_nchw(const u16* __restrict__ A, const u16* __restrict__ Bm,
                    const float* __restrict__ bias, u16* __restrict__ Out,
                    int M, int N, int K, int SP)
{
  __shared__ __align__(16) u16 As[2][128 * KTP];
  __shared__ __align__(16) u16 Bs[2][128 * KTP];

  const int tid  = threadIdx.x;
  const int lane = tid & 31;
  const int wave = tid >> 5;
  const int wm   = wave & 1;        // 0..1 -> 64 rows each
  const int wn   = wave >> 1;       // 0..3 -> 32 cols each
  const int l15  = lane & 15;
  const int hi   = (lane >> 4) & 1;
  const int m0   = blockIdx.y * 128;
  const int n0   = blockIdx.x * 128;

  const int srow = tid >> 1;         // staging row 0..127
  const int scol = (tid & 1) << 4;   // staging col 0 or 16

  // clamp tail rows: garbage A rows only feed C rows >= M (never stored)
  int grA = m0 + srow; if (grA > M - 1) grA = M - 1;
  const int grB = n0 + srow;

  v8f acc[4][2];
  const v8f vzero = {0.f,0.f,0.f,0.f,0.f,0.f,0.f,0.f};
  for (int i = 0; i < 4; ++i)
    for (int j = 0; j < 2; ++j)
      acc[i][j] = vzero;

  // 4 async DMA ops per thread per stage: A(0,16) + B(0,16); IOFFSET applies to
  // both the LDS and global address (ISA 15.18 GLOBAL_LOAD_ASYNC_TO_LDS).
  auto stage = [&](int kk, int buf) {
    const u16* gA = A  + (size_t)grA * K + kk + scol;
    const u16* gB = Bm + (size_t)grB * K + kk + scol;
    u32 ldsA = (u32)(size_t)&As[buf][srow * KTP + scol];
    u32 ldsB = (u32)(size_t)&Bs[buf][srow * KTP + scol];
    asm volatile(
      "global_load_async_to_lds_b128 %0, %1, off\n\t"
      "global_load_async_to_lds_b128 %0, %1, off offset:16"
      :: "v"(ldsA), "v"(gA) : "memory");
    asm volatile(
      "global_load_async_to_lds_b128 %0, %1, off\n\t"
      "global_load_async_to_lds_b128 %0, %1, off offset:16"
      :: "v"(ldsB), "v"(gB) : "memory");
    if (kk + 64 < K)
      __builtin_prefetch(gA + 64, 0, 1);   // global_prefetch_b8, two tiles ahead
  };

  stage(0, 0);

  const int nk = K >> 5;
  for (int ik = 0; ik < nk; ++ik) {
    const int buf = ik & 1;
    if (ik + 1 < nk) {
      stage((ik + 1) << 5, buf ^ 1);
      WAIT_ASYNC(4);       // drain current tile's 4 ops, keep next tile in flight
    } else {
      WAIT_ASYNC(0);
    }
    __syncthreads();       // all waves' DMA for `buf` complete -> safe to read

    Frag16 fga[4], fgb[2];
    const int ka = hi ? 8 : 0;   // A lane layout: K {0..7,16..23} | {8..15,24..31}
    for (int mi = 0; mi < 4; ++mi) {
      const u16* p = &As[buf][(wm * 64 + mi * 16 + l15) * KTP + ka];
      fga[mi].q[0] = *(const uint4*)p;
      fga[mi].q[1] = *(const uint4*)(p + 16);
    }
    const int kb = hi ? 8 : 0;   // B lane layout: K {0..7,16..23} | {8..15,24..31}
    for (int ni = 0; ni < 2; ++ni) {
      const u16* p = &Bs[buf][(wn * 32 + ni * 16 + l15) * KTP + kb];
      fgb[ni].q[0] = *(const uint4*)p;
      fgb[ni].q[1] = *(const uint4*)(p + 16);
    }
    for (int mi = 0; mi < 4; ++mi)
      for (int ni = 0; ni < 2; ++ni)
        acc[mi][ni] = __builtin_amdgcn_wmma_f32_16x16x32_bf16(
            false, fga[mi].v, false, fgb[ni].v, (short)0, acc[mi][ni], false, false);
    __syncthreads();       // done reading `buf` before next stage overwrites it
  }

  // epilogue: bias + relu, scatter into NCHW bf16; one div per 8-row run,
  // sp is contiguous per lane so stores within a lane hit the same lines.
  for (int mi = 0; mi < 4; ++mi) {
    const int rbase = m0 + wm * 64 + mi * 16 + hi * 8;
    const int bb  = rbase / SP;
    const int sp0 = rbase - bb * SP;
    for (int ni = 0; ni < 2; ++ni) {
      const int col  = n0 + wn * 32 + ni * 16 + l15;
      const float bv = bias[col];
      v8f c = acc[mi][ni];
      int b = bb, sp = sp0;
      for (int r = 0; r < 8; ++r) {
        if (rbase + r < M) {
          float v = c[r] + bv;
          v = v > 0.f ? v : 0.f;
          Out[((size_t)b * N + col) * SP + sp] = f2bf(v);
        }
        if (++sp == SP) { sp = 0; ++b; }
      }
    }
  }
}

// ---------------------------------------------------------------------------
// mean over spatial: feat[b*C+c] = mean_sp O3
// ---------------------------------------------------------------------------
__global__ void meanpool_kernel(const u16* __restrict__ O, float* __restrict__ feat,
                                int BC, int SP)
{
  int idx = blockIdx.x * blockDim.x + threadIdx.x;
  if (idx >= BC) return;
  const u16* p = O + (size_t)idx * SP;
  float s = 0.f;
  for (int i = 0; i < SP; ++i) s += bf2f(p[i]);
  feat[idx] = s * (1.f / (float)SP);
}

// fourier features: fvec[b][v*20 + f] = fa*sin, [+10+f] = fa*cos
__global__ void fourier_kernel(const float* __restrict__ dyn, const float* __restrict__ goal,
                               const float* __restrict__ fa, const float* __restrict__ fb,
                               float* __restrict__ fvec)
{
  int idx = blockIdx.x * blockDim.x + threadIdx.x;  // 64*4*10
  if (idx >= 64 * 4 * 10) return;
  int f = idx % 10;
  int v = (idx / 10) & 3;
  int b = idx / 40;
  float x = (v < 2) ? dyn[b * 2 + v] : goal[b * 2 + v - 2];
  float t = 6.283185307179586f * fb[f] * x;
  float s, c;  __sincosf(t, &s, &c);
  fvec[b * 80 + v * 20 + f]      = fa[f] * s;
  fvec[b * 80 + v * 20 + 10 + f] = fa[f] * c;
}

__global__ void enc1_kernel(const float* __restrict__ feat, const float* __restrict__ fvec,
                            const float* __restrict__ w1, const float* __restrict__ b1,
                            float* __restrict__ e)
{
  int idx = blockIdx.x * blockDim.x + threadIdx.x;   // 64*512
  if (idx >= 64 * 512) return;
  int j = idx & 511, b = idx >> 9;
  float s = b1[j];
  const float* fb_ = feat + (size_t)b * 2048;
  for (int k = 0; k < 2048; ++k) s += fb_[k] * w1[(size_t)k * 512 + j];
  const float* fv = fvec + b * 80;
  for (int k = 0; k < 80; ++k)   s += fv[k] * w1[(size_t)(2048 + k) * 512 + j];
  e[idx] = tanhf(s);
}

__global__ void enc2_kernel(const float* __restrict__ e, const float* __restrict__ w2,
                            const float* __restrict__ b2, float* __restrict__ h)
{
  int idx = blockIdx.x * blockDim.x + threadIdx.x;   // 64*64
  if (idx >= 64 * 64) return;
  int j = idx & 63, b = idx >> 6;
  float s = b2[j];
  const float* eb = e + b * 512;
  for (int k = 0; k < 512; ++k) s += eb[k] * w2[k * 64 + j];
  h[idx] = s;
}

// ---------------------------------------------------------------------------
// Persistent GRU rollout: 1 block x 1024 threads, T=75 steps, H=B=64, ADIM=2.
// Hidden states rotate through 3 LDS buffers; acts[t] = a before update.
// ---------------------------------------------------------------------------
__device__ __forceinline__ float sigf(float x) { return 1.f / (1.f + __expf(-x)); }

__global__ __launch_bounds__(1024)
void gru_rollout_kernel(const float* __restrict__ hinit, const float* __restrict__ cmd,
                        const float* __restrict__ wi0, const float* __restrict__ wh0,
                        const float* __restrict__ bi0, const float* __restrict__ bh0,
                        const float* __restrict__ wi1, const float* __restrict__ wh1,
                        const float* __restrict__ bi1, const float* __restrict__ bh1,
                        const float* __restrict__ dw1, const float* __restrict__ db1,
                        const float* __restrict__ dw2, const float* __restrict__ db2,
                        float* __restrict__ out)
{
  const int T = 75;
  __shared__ float bufs[3][64 * 64];
  __shared__ float aL[128];
  __shared__ float anL[128];
  const int tid = threadIdx.x;

  float* h0  = bufs[0];
  float* h1  = bufs[1];
  float* tmp = bufs[2];
  for (int i = tid; i < 64 * 64; i += 1024) { h0[i] = hinit[i]; h1[i] = hinit[i]; }
  if (tid < 128) aL[tid] = cmd[(tid >> 1) * T * 2 + (tid & 1)];
  __syncthreads();

  for (int t = 0; t < T; ++t) {
    if (tid < 128) out[(tid >> 1) * T * 2 + t * 2 + (tid & 1)] = aL[tid];

    // cell 0: x = a (dim 2), h = h0 -> tmp
    for (int p = tid; p < 4096; p += 1024) {
      const int b = p >> 6, j = p & 63;
      const float a0 = aL[b * 2], a1 = aL[b * 2 + 1];
      float ir = bi0[j]       + wi0[j * 2] * a0         + wi0[j * 2 + 1] * a1;
      float iz = bi0[64 + j]  + wi0[(64 + j) * 2] * a0  + wi0[(64 + j) * 2 + 1] * a1;
      float in_= bi0[128 + j] + wi0[(128 + j) * 2] * a0 + wi0[(128 + j) * 2 + 1] * a1;
      float hr = bh0[j], hz = bh0[64 + j], hn = bh0[128 + j];
      const float* hb = &h0[b * 64];
      for (int k = 0; k < 64; ++k) {
        const float hv = hb[k];
        hr += wh0[j * 64 + k] * hv;
        hz += wh0[(64 + j) * 64 + k] * hv;
        hn += wh0[(128 + j) * 64 + k] * hv;
      }
      const float r = sigf(ir + hr);
      const float z = sigf(iz + hz);
      const float n = tanhf(in_ + r * hn);
      tmp[b * 64 + j] = (1.f - z) * n + z * hb[j];
    }
    __syncthreads();

    // cell 1: x = tmp (h0n), h = h1 -> write into old h0 storage
    for (int p = tid; p < 4096; p += 1024) {
      const int b = p >> 6, j = p & 63;
      float ir = bi1[j], iz = bi1[64 + j], in_ = bi1[128 + j];
      float hr = bh1[j], hz = bh1[64 + j], hn  = bh1[128 + j];
      const float* xb = &tmp[b * 64];
      const float* hb = &h1[b * 64];
      for (int k = 0; k < 64; ++k) {
        const float xv = xb[k];
        ir += wi1[j * 64 + k] * xv;
        iz += wi1[(64 + j) * 64 + k] * xv;
        in_+= wi1[(128 + j) * 64 + k] * xv;
        const float hv = hb[k];
        hr += wh1[j * 64 + k] * hv;
        hz += wh1[(64 + j) * 64 + k] * hv;
        hn += wh1[(128 + j) * 64 + k] * hv;
      }
      const float r = sigf(ir + hr);
      const float z = sigf(iz + hz);
      const float n = tanhf(in_ + r * hn);
      h0[b * 64 + j] = (1.f - z) * n + z * hb[j];   // h1n
    }
    __syncthreads();

    // decoder: act = tanh(tanh(h1n @ dw1 + db1) @ dw2 + db2)
    if (tid < 64) {
      const int b = tid;
      const float* hb = &h0[b * 64];   // h1n
      float d16[16];
      for (int i = 0; i < 16; ++i) {
        float s = db1[i];
        for (int k = 0; k < 64; ++k) s += hb[k] * dw1[k * 16 + i];
        d16[i] = tanhf(s);
      }
      for (int d = 0; d < 2; ++d) {
        float s = db2[d];
        for (int i = 0; i < 16; ++i) s += d16[i] * dw2[i * 2 + d];
        anL[b * 2 + d] = tanhf(s);
      }
    }
    __syncthreads();
    if (tid < 128) aL[tid] = anL[tid];
    // rotate: h0' = tmp (h0n), h1' = old h0 (holds h1n), tmp' = old h1 (dead)
    float* nh0 = tmp; float* nh1 = h0; float* ntm = h1;
    h0 = nh0; h1 = nh1; tmp = ntm;
    __syncthreads();
  }
}

// ---------------------------------------------------------------------------
extern "C" void kernel_launch(void* const* d_in, const int* in_sizes, int n_in,
                              void* d_out, int out_size, void* d_ws, size_t ws_size,
                              hipStream_t stream)
{
  const float* image = (const float*)d_in[0];
  const float* dyn   = (const float*)d_in[1];
  const float* goal  = (const float*)d_in[2];
  const float* cmd   = (const float*)d_in[3];
  const float* c1w = (const float*)d_in[4];  const float* c1b = (const float*)d_in[5];
  const float* c2w = (const float*)d_in[6];  const float* c2b = (const float*)d_in[7];
  const float* c3w = (const float*)d_in[8];  const float* c3b = (const float*)d_in[9];
  const float* fa  = (const float*)d_in[10]; const float* fbv = (const float*)d_in[11];
  const float* ew1 = (const float*)d_in[12]; const float* eb1 = (const float*)d_in[13];
  const float* ew2 = (const float*)d_in[14]; const float* eb2 = (const float*)d_in[15];
  const float* wi0 = (const float*)d_in[16]; const float* wh0 = (const float*)d_in[17];
  const float* bi0 = (const float*)d_in[18]; const float* bh0 = (const float*)d_in[19];
  const float* wi1 = (const float*)d_in[20]; const float* wh1 = (const float*)d_in[21];
  const float* bi1 = (const float*)d_in[22]; const float* bh1 = (const float*)d_in[23];
  const float* dw1 = (const float*)d_in[24]; const float* db1 = (const float*)d_in[25];
  const float* dw2 = (const float*)d_in[26]; const float* db2 = (const float*)d_in[27];

  // workspace carve-up (A-slot and B-slot reused across the three convs)
  char* w = (char*)d_ws;
  auto carve = [&](size_t bytes) -> char* {
    char* p = w; w += (bytes + 255) & ~(size_t)255; return p;
  };
  u16*   Aslot = (u16*)carve((size_t)238144 * 160 * 2);       // 76.2 MB (max im2col)
  u16*   Bslot = (u16*)carve((size_t)2048 * 9216 * 2);        // 37.7 MB (max packed W)
  u16*   O1    = (u16*)carve((size_t)64 * 256 * 3721 * 2);    // conv1 out bf16
  u16*   O2    = (u16*)carve((size_t)64 * 1024 * 256 * 2);    // conv2 out bf16
  u16*   O3    = (u16*)carve((size_t)64 * 2048 * 64 * 2);     // conv3 out bf16
  float* feat  = (float*)carve((size_t)64 * 2048 * 4);
  float* fvec  = (float*)carve((size_t)64 * 80 * 4);
  float* ebuf  = (float*)carve((size_t)64 * 512 * 4);
  float* hbuf  = (float*)carve((size_t)64 * 64 * 4);

  const int TPB = 256;
  auto blocks = [](long long n) { return (unsigned)((n + 255) / 256); };

  // conv1: M=238144 (64*61*61), N=256, K=147 -> pad 160, SAME pad_lo=1, stride 4
  {
    long long wtot = 256LL * 160;
    pack_w_kernel<<<blocks(wtot), TPB, 0, stream>>>(c1w, Bslot, 147, 160, wtot);
    long long atot = 238144LL * 160;
    im2col_kernel<1><<<blocks(atot), TPB, 0, stream>>>(image, Aslot, 3, 244, 244,
                                                       61, 61, 7, 4, 1, 147, 160, atot);
    dim3 g(256 / 128, (238144 + 127) / 128);
    wmma_gemm_nchw<<<g, TPB, 0, stream>>>(Aslot, Bslot, c1b, O1, 238144, 256, 160, 3721);
  }
  // conv2: M=16384 (64*16*16), N=1024, K=2304, SAME pad_lo=1, stride 4
  {
    long long wtot = 1024LL * 2304;
    pack_w_kernel<<<blocks(wtot), TPB, 0, stream>>>(c2w, Bslot, 2304, 2304, wtot);
    long long atot = 16384LL * 2304;
    im2col_kernel<0><<<blocks(atot), TPB, 0, stream>>>(O1, Aslot, 256, 61, 61,
                                                       16, 16, 3, 4, 1, 2304, 2304, atot);
    dim3 g(1024 / 128, 16384 / 128);
    wmma_gemm_nchw<<<g, TPB, 0, stream>>>(Aslot, Bslot, c2b, O2, 16384, 1024, 2304, 256);
  }
  // conv3: M=4096 (64*8*8), N=2048, K=9216, SAME pad_lo=0, stride 2
  {
    long long wtot = 2048LL * 9216;
    pack_w_kernel<<<blocks(wtot), TPB, 0, stream>>>(c3w, Bslot, 9216, 9216, wtot);
    long long atot = 4096LL * 9216;
    im2col_kernel<0><<<blocks(atot), TPB, 0, stream>>>(O2, Aslot, 1024, 16, 16,
                                                       8, 8, 3, 2, 0, 9216, 9216, atot);
    dim3 g(2048 / 128, 4096 / 128);
    wmma_gemm_nchw<<<g, TPB, 0, stream>>>(Aslot, Bslot, c3b, O3, 4096, 2048, 9216, 64);
  }
  // pooled features + fourier + encoder
  meanpool_kernel<<<blocks(64 * 2048), TPB, 0, stream>>>(O3, feat, 64 * 2048, 64);
  fourier_kernel<<<blocks(64 * 4 * 10), TPB, 0, stream>>>(dyn, goal, fa, fbv, fvec);
  enc1_kernel<<<blocks(64 * 512), TPB, 0, stream>>>(feat, fvec, ew1, eb1, ebuf);
  enc2_kernel<<<blocks(64 * 64), TPB, 0, stream>>>(ebuf, ew2, eb2, hbuf);
  // sequential GRU rollout (persistent single workgroup)
  gru_rollout_kernel<<<1, 1024, 0, stream>>>(hbuf, cmd, wi0, wh0, bi0, bh0,
                                             wi1, wh1, bi1, bh1,
                                             dw1, db1, dw2, db2, (float*)d_out);
}
